// GQA_10548439679158
// MI455X (gfx1250) — matrix operand order repaired and checked by reading, not compiled, not run on hardware
//
#include <hip/hip_runtime.h>
#include <hip/hip_bf16.h>
#include <cstdint>

// ---------------------------------------------------------------------------
// CDNA5 (gfx1250) GQA block: cvt->bf16, QKV proj, RoPE, flash attn, out proj.
// WMMA bf16 (f32 acc) everywhere matmul-shaped; async global->LDS staging with
// double buffering (ASYNCcnt), ds_load_b128 fragment loads.
// ---------------------------------------------------------------------------

typedef __attribute__((ext_vector_type(16))) __bf16 v16bf;
typedef __attribute__((ext_vector_type(8)))  float  v8f;

union FragU { v16bf v; unsigned int u[8]; };

#define S_LEN   2048
#define DMODEL  4096
#define DKV     1024
#define HQ      32
#define HKV     8
#define HD      128

// ---------------- bf16 conversion helpers ----------------
__device__ __forceinline__ unsigned short f32_to_bf16(float f) {
    unsigned int u = __float_as_uint(f);
    unsigned int r = u + 0x7FFFu + ((u >> 16) & 1u);   // RNE
    return (unsigned short)(r >> 16);
}

__device__ __forceinline__ unsigned int pack_bf16x2(float lo, float hi) {
#if __has_builtin(__builtin_amdgcn_cvt_pk_bf16_f32)
    auto r = __builtin_amdgcn_cvt_pk_bf16_f32(lo, hi);
    unsigned int u;
    __builtin_memcpy(&u, &r, 4);
    return u;
#else
    return (unsigned int)f32_to_bf16(lo) | ((unsigned int)f32_to_bf16(hi) << 16);
#endif
}

// ---------------- async global -> LDS (16B per lane) ----------------
#if __has_builtin(__builtin_amdgcn_global_load_async_to_lds_b128)
#define HAVE_ASYNC_LDS 1
#endif

// Builtin expects GCC-vector int4 pointers: global (AS1 = __device__) src,
// LDS (AS3 = __shared__) dst.  (Exact param type taken from hipcc diagnostic.)
typedef int v4i_gcc __attribute__((vector_size(16)));
typedef __attribute__((address_space(1))) v4i_gcc glb_v4i_t;
typedef __attribute__((address_space(3))) v4i_gcc lds_v4i_t;

__device__ __forceinline__ void copy16_g2l(unsigned short* lds_dst,
                                           const unsigned short* gsrc) {
#ifdef HAVE_ASYNC_LDS
    __builtin_amdgcn_global_load_async_to_lds_b128(
        (glb_v4i_t*)gsrc, (lds_v4i_t*)lds_dst, 0, 0);
#else
    *(uint4*)lds_dst = *(const uint4*)gsrc;
#endif
}

__device__ __forceinline__ void wait_async_all() {
#ifdef HAVE_ASYNC_LDS
#if __has_builtin(__builtin_amdgcn_s_wait_asynccnt)
    __builtin_amdgcn_s_wait_asynccnt(0);
#else
    asm volatile("s_wait_asynccnt 0x0" ::: "memory");
#endif
#endif
}

// ---------------- WMMA fragment assembly (contiguous k order, same for A and B) ----------------
// A (16x32, MxK): lane holds M=lane&15; VGPR p covers
//   kstart = ((lane>=16)?16:0) + 2p
__device__ __forceinline__ v16bf load_frag_a(const unsigned short* base,
                                             int row0, int strideUS, int lane) {
    FragU f;
    const int m = row0 + (lane & 15);
    const int klane = (lane >> 4) << 4;
    const unsigned int* p32 = (const unsigned int*)base;
    const int rowoff = m * strideUS;
#pragma unroll
    for (int p = 0; p < 8; ++p) {
        int kst = klane + (p << 1);
        f.u[p] = p32[(rowoff + kst) >> 1];
    }
    return f.v;
}

// B (32x16, KxN) from N-major rows of K: lane holds N=lane&15;
//   VGPR p: kstart = ((lane>=16)?16:0) + 2p
__device__ __forceinline__ v16bf load_frag_b_nk(const unsigned short* base,
                                                int col0, int strideUS, int lane) {
    FragU f;
    const int n = col0 + (lane & 15);
    const int kb = (lane >> 4) << 4;
    const unsigned int* p32 = (const unsigned int*)base;
    const int rowoff = n * strideUS;
#pragma unroll
    for (int p = 0; p < 8; ++p) {
        f.u[p] = p32[(rowoff + kb + (p << 1)) >> 1];
    }
    return f.v;
}

// ---------------------------------------------------------------------------
// f32 -> bf16 bulk convert (memory-bound, one pass per tensor)
// ---------------------------------------------------------------------------
__global__ __launch_bounds__(256) void cvt_f32_to_bf16(
    const float* __restrict__ src, unsigned short* __restrict__ dst, int n4)
{
    int i = blockIdx.x * blockDim.x + threadIdx.x;
    const int stride = gridDim.x * blockDim.x;
    for (; i < n4; i += stride) {
        float4 v = ((const float4*)src)[i];
        uint2 o;
        o.x = pack_bf16x2(v.x, v.y);
        o.y = pack_bf16x2(v.z, v.w);
        ((uint2*)dst)[i] = o;
    }
}

// ---------------------------------------------------------------------------
// GEMM: C(MxN,f32) = A(MxK,bf16) @ B(NxK,bf16)^T
// 128x128 block tile, 256 threads (8 waves), wave tile 32x64,
// double-buffered LDS with async global->LDS staging.
// ---------------------------------------------------------------------------
__global__ __launch_bounds__(256) void gemm_bf16_nt(
    const unsigned short* __restrict__ A, const unsigned short* __restrict__ B,
    float* __restrict__ C, int M, int N, int K)
{
    __shared__ unsigned short sA[2][128 * 32];
    __shared__ unsigned short sB[2][128 * 32];

    const int tid  = threadIdx.x;
    const int lane = tid & 31;
    const int w    = tid >> 5;
    const int m_blk = blockIdx.y * 128;
    const int n_blk = blockIdx.x * 128;
    const int m_base = (w >> 1) * 32;
    const int n_base = (w & 1) * 64;

    v8f acc[2][4];
#pragma unroll
    for (int i = 0; i < 2; ++i)
#pragma unroll
        for (int j = 0; j < 4; ++j) acc[i][j] = (v8f){};

    auto stage = [&](int buf, int k0) {
#pragma unroll
        for (int it = 0; it < 2; ++it) {
            int idx = tid + it * 256;       // 512 x 16B chunks per matrix
            int row = idx >> 2;             // 0..127
            int c16 = (idx & 3) << 3;       // bf16 col 0,8,16,24
            copy16_g2l(&sA[buf][row * 32 + c16],
                       A + (size_t)(m_blk + row) * K + k0 + c16);
            copy16_g2l(&sB[buf][row * 32 + c16],
                       B + (size_t)(n_blk + row) * K + k0 + c16);
        }
    };

    stage(0, 0);
    wait_async_all();
    __syncthreads();

    int buf = 0;
    for (int k0 = 0; k0 < K; k0 += 32) {
        if (k0 + 32 < K) stage(buf ^ 1, k0 + 32);   // prefetch next tile

        const unsigned short* at = &sA[buf][0];
        const unsigned short* bt = &sB[buf][0];
        v16bf afrag[2];
        afrag[0] = load_frag_a(at, m_base,      32, lane);
        afrag[1] = load_frag_a(at, m_base + 16, 32, lane);
#pragma unroll
        for (int nt = 0; nt < 4; ++nt) {
            v16bf bfrag = load_frag_b_nk(bt, n_base + nt * 16, 32, lane);
            acc[0][nt] = __builtin_amdgcn_wmma_f32_16x16x32_bf16(
                false, afrag[0], false, bfrag, (short)0, acc[0][nt], false, false);
            acc[1][nt] = __builtin_amdgcn_wmma_f32_16x16x32_bf16(
                false, afrag[1], false, bfrag, (short)0, acc[1][nt], false, false);
        }
        wait_async_all();
        __syncthreads();
        buf ^= 1;
    }

    const int mrow_add = (lane >> 4) << 3;
    const int ncol     = lane & 15;
#pragma unroll
    for (int i = 0; i < 2; ++i)
#pragma unroll
        for (int j = 0; j < 4; ++j) {
            int cbase = n_blk + n_base + j * 16 + ncol;
#pragma unroll
            for (int e = 0; e < 8; ++e) {
                int r = m_blk + m_base + i * 16 + mrow_add + e;
                C[(size_t)r * N + cbase] = acc[i][j][e];
            }
        }
}

// ---------------------------------------------------------------------------
// RoPE + head transpose + bf16 pack. grid = (S_LEN, 48)
// ---------------------------------------------------------------------------
__global__ __launch_bounds__(128) void rope_pack_kernel(
    const float* __restrict__ qf, const float* __restrict__ kf,
    const float* __restrict__ vf, const float* __restrict__ freq,
    unsigned short* __restrict__ q_bf, unsigned short* __restrict__ k_bf,
    unsigned short* __restrict__ v_bf,
    float* __restrict__ out_k, float* __restrict__ out_v)
{
    const int s = blockIdx.x;
    const int y = blockIdx.y;
    const int i = threadIdx.x;

    const float f  = freq[s * HD + i];
    const float cf = __cosf(f);
    const float sf = __sinf(f);

    if (y < HQ) {
        const float* row = qf + (size_t)s * DMODEL + y * HD;
        float v   = row[i];
        float rot = (i < HD / 2) ? -row[i + HD / 2] : row[i - HD / 2];
        float r   = v * cf + rot * sf;
        q_bf[((size_t)y * S_LEN + s) * HD + i] = f32_to_bf16(r);
    } else if (y < HQ + HKV) {
        const int h = y - HQ;
        const float* row = kf + (size_t)s * DKV + h * HD;
        float v   = row[i];
        float rot = (i < HD / 2) ? -row[i + HD / 2] : row[i - HD / 2];
        float r   = v * cf + rot * sf;
        size_t o  = ((size_t)h * S_LEN + s) * HD + i;
        k_bf[o]  = f32_to_bf16(r);
        out_k[o] = r;
    } else {
        const int h = y - HQ - HKV;
        float v  = vf[(size_t)s * DKV + h * HD + i];
        size_t o = ((size_t)h * S_LEN + s) * HD + i;
        v_bf[o]  = f32_to_bf16(v);
        out_v[o] = v;
    }
}

// ---------------------------------------------------------------------------
// Flash attention (causal, GQA 4:1). grid = (S_LEN/64, HQ), 128 threads.
// Writes attention output directly as bf16 for the final projection.
// ---------------------------------------------------------------------------
__global__ __launch_bounds__(128) void flash_attn_kernel(
    const unsigned short* __restrict__ q_bf,
    const unsigned short* __restrict__ k_bf,
    const unsigned short* __restrict__ v_bf,
    unsigned short* __restrict__ attn_bf)   // (S_LEN, HQ*HD) bf16
{
    __shared__ unsigned short sK [32 * HD];    // (kc, d)
    __shared__ unsigned short sVt[HD * 32];    // transposed (d, kc)
    __shared__ unsigned short sP [4][16 * 32];

    const int h     = blockIdx.y;
    const int hkv   = h >> 2;
    const int q0    = blockIdx.x * 64;
    const int tid   = threadIdx.x;
    const int lane  = tid & 31;
    const int w     = tid >> 5;
    const int qrow0 = q0 + w * 16;

    // Q fragments resident in registers (16 rows x 128 dims = 4 K-chunks)
    v16bf qfrag[4];
    {
        const int m     = lane & 15;
        const int klane = (lane >> 4) << 4;
        const unsigned int* qp = (const unsigned int*)q_bf;
        const size_t rowbase = ((size_t)h * S_LEN + qrow0 + m) * HD;
#pragma unroll
        for (int c = 0; c < 4; ++c) {
            FragU fr;
#pragma unroll
            for (int p = 0; p < 8; ++p) {
                int kst = c * 32 + klane + (p << 1);
                fr.u[p] = qp[(rowbase + kst) >> 1];
            }
            qfrag[c] = fr.v;
        }
    }

    float m_run[8], l_run[8];
    v8f   o_acc[8];
#pragma unroll
    for (int j = 0; j < 8; ++j) { m_run[j] = -3.0e38f; l_run[j] = 0.f; }
#pragma unroll
    for (int dt = 0; dt < 8; ++dt) o_acc[dt] = (v8f){};

    const float scale = 0.08838834764831845f;   // 1/sqrt(128)
    const int kt_end  = (q0 + 64) >> 5;

    for (int kt = 0; kt < kt_end; ++kt) {
        const unsigned short* kb = k_bf + ((size_t)hkv * S_LEN + kt * 32) * HD;
        const unsigned short* vb = v_bf + ((size_t)hkv * S_LEN + kt * 32) * HD;
        // K tile: async copy, row-major
#pragma unroll
        for (int it = 0; it < 4; ++it) {
            int idx = tid + it * 128;           // 512 x 16B
            int kc  = idx >> 4;
            int d0  = (idx & 15) << 3;
            copy16_g2l(&sK[kc * HD + d0], kb + kc * HD + d0);
        }
        // V tile: packed load + transpose scatter into (d, kc)
#pragma unroll
        for (int it = 0; it < 4; ++it) {
            int idx = tid + it * 128;
            int kc  = idx >> 4;
            int d0  = (idx & 15) << 3;
            uint4 vv = *(const uint4*)(vb + kc * HD + d0);
            const unsigned short* pe = (const unsigned short*)&vv;
#pragma unroll
            for (int j = 0; j < 8; ++j)
                sVt[(d0 + j) * 32 + kc] = pe[j];
        }
        if (kt + 1 < kt_end)
            __builtin_prefetch(kb + 32 * HD + tid * 8, 0, 1);
        wait_async_all();
        __syncthreads();

        // S = Q . K^T (16x32 per wave)
        v8f sacc[2];
        sacc[0] = (v8f){}; sacc[1] = (v8f){};
#pragma unroll
        for (int nt = 0; nt < 2; ++nt) {
            const int n   = nt * 16 + (lane & 15);
            const int kb2 = (lane >> 4) << 4;
#pragma unroll
            for (int c = 0; c < 4; ++c) {
                FragU fb;
#pragma unroll
                for (int p = 0; p < 8; ++p) {
                    fb.u[p] = ((const unsigned int*)sK)[(n * HD + c * 32 + kb2 + (p << 1)) >> 1];
                }
                sacc[nt] = __builtin_amdgcn_wmma_f32_16x16x32_bf16(
                    false, qfrag[c], false, fb.v, (short)0, sacc[nt], false, false);
            }
        }

        // scale + causal mask + online softmax
        const int rbase = qrow0 + ((lane >> 4) << 3);
        const int cbase = kt * 32 + (lane & 15);
#pragma unroll
        for (int j = 0; j < 8; ++j) {
            const int r = rbase + j;
            float s0 = (cbase      <= r) ? sacc[0][j] * scale : -3.0e38f;
            float s1 = (cbase + 16 <= r) ? sacc[1][j] * scale : -3.0e38f;

            float mx = fmaxf(s0, s1);
#pragma unroll
            for (int off = 1; off < 16; off <<= 1)
                mx = fmaxf(mx, __shfl_xor(mx, off, 32));
            float mnew = fmaxf(m_run[j], mx);
            float corr = __expf(m_run[j] - mnew);
            m_run[j] = mnew;

            float p0 = __expf(s0 - mnew);
            float p1 = __expf(s1 - mnew);
            float rs = p0 + p1;
#pragma unroll
            for (int off = 1; off < 16; off <<= 1)
                rs += __shfl_xor(rs, off, 32);
            l_run[j] = l_run[j] * corr + rs;

#pragma unroll
            for (int dt = 0; dt < 8; ++dt)
                o_acc[dt][j] = o_acc[dt][j] * corr;

            const int r_loc = j + ((lane >> 4) << 3);
            const int c_loc = lane & 15;
            sP[w][r_loc * 32 + c_loc]      = f32_to_bf16(p0);
            sP[w][r_loc * 32 + c_loc + 16] = f32_to_bf16(p1);
        }
        asm volatile("s_wait_dscnt 0x0" ::: "memory");  // wave-local LDS RAW

        // O += P . V
        FragU fa;
        {
            const int m     = lane & 15;
            const int klane = (lane >> 4) << 4;
            const unsigned int* pp = (const unsigned int*)(&sP[w][0]);
#pragma unroll
            for (int p = 0; p < 8; ++p) {
                int kst = klane + (p << 1);
                fa.u[p] = pp[(m * 32 + kst) >> 1];
            }
        }
#pragma unroll
        for (int dt = 0; dt < 8; ++dt) {
            FragU fb;
            const int n   = dt * 16 + (lane & 15);
            const int kb2 = (lane >> 4) << 4;
#pragma unroll
            for (int p = 0; p < 8; ++p) {
                fb.u[p] = ((const unsigned int*)sVt)[(n * 32 + kb2 + (p << 1)) >> 1];
            }
            o_acc[dt] = __builtin_amdgcn_wmma_f32_16x16x32_bf16(
                false, fa.v, false, fb.v, (short)0, o_acc[dt], false, false);
        }
        __syncthreads();
    }

    // finalize: O /= l, emit bf16 directly for output projection
    {
        const int mrow = (lane >> 4) << 3;
        const int ncol = lane & 15;
#pragma unroll
        for (int j = 0; j < 8; ++j) {
            float inv = 1.0f / l_run[j];
            int r = qrow0 + mrow + j;
#pragma unroll
            for (int dt = 0; dt < 8; ++dt) {
                attn_bf[(size_t)r * DMODEL + h * HD + dt * 16 + ncol] =
                    f32_to_bf16(o_acc[dt][j] * inv);
            }
        }
    }
}

// ---------------------------------------------------------------------------
// Launch
// ---------------------------------------------------------------------------
extern "C" void kernel_launch(void* const* d_in, const int* in_sizes, int n_in,
                              void* d_out, int out_size, void* d_ws, size_t ws_size,
                              hipStream_t stream) {
    const float* x    = (const float*)d_in[0];
    const float* freq = (const float*)d_in[1];
    const float* wq   = (const float*)d_in[2];
    const float* wk   = (const float*)d_in[3];
    const float* wv   = (const float*)d_in[4];
    const float* wo   = (const float*)d_in[5];
    float* out = (float*)d_out;

    char* ws = (char*)d_ws;
    float* qf = (float*)ws;                      ws += (size_t)S_LEN * DMODEL * 4; // 32 MB
    float* kf = (float*)ws;                      ws += (size_t)S_LEN * DKV * 4;    //  8 MB
    float* vf = (float*)ws;                      ws += (size_t)S_LEN * DKV * 4;    //  8 MB
    unsigned short* x_bf  = (unsigned short*)ws; ws += (size_t)S_LEN * DMODEL * 2; // 16 MB
    unsigned short* wq_bf = (unsigned short*)ws; ws += (size_t)DMODEL * DMODEL * 2;// 32 MB
    unsigned short* wk_bf = (unsigned short*)ws; ws += (size_t)DKV * DMODEL * 2;   //  8 MB
    unsigned short* wv_bf = (unsigned short*)ws; ws += (size_t)DKV * DMODEL * 2;   //  8 MB
    unsigned short* wo_bf = (unsigned short*)ws; ws += (size_t)DMODEL * DMODEL * 2;// 32 MB
    unsigned short* q_bf  = (unsigned short*)ws; ws += (size_t)HQ  * S_LEN * HD * 2;// 16 MB
    unsigned short* k_bf  = (unsigned short*)ws; ws += (size_t)HKV * S_LEN * HD * 2;//  4 MB
    unsigned short* v_bf  = (unsigned short*)ws; ws += (size_t)HKV * S_LEN * HD * 2;//  4 MB
    unsigned short* attn_bf = (unsigned short*)ws; ws += (size_t)S_LEN * DMODEL * 2;// 16 MB

    const dim3 blk256(256);
    const dim3 cvtgrid(1024);

    // 0) one-time f32 -> bf16 conversions (memory-bound)
    cvt_f32_to_bf16<<<cvtgrid, blk256, 0, stream>>>(x,  x_bf,  S_LEN * DMODEL / 4);
    cvt_f32_to_bf16<<<cvtgrid, blk256, 0, stream>>>(wq, wq_bf, DMODEL * DMODEL / 4);
    cvt_f32_to_bf16<<<cvtgrid, blk256, 0, stream>>>(wk, wk_bf, DKV * DMODEL / 4);
    cvt_f32_to_bf16<<<cvtgrid, blk256, 0, stream>>>(wv, wv_bf, DKV * DMODEL / 4);
    cvt_f32_to_bf16<<<cvtgrid, blk256, 0, stream>>>(wo, wo_bf, DMODEL * DMODEL / 4);

    // 1) projections
    gemm_bf16_nt<<<dim3(DMODEL / 128, S_LEN / 128), blk256, 0, stream>>>(
        x_bf, wq_bf, qf, S_LEN, DMODEL, DMODEL);
    gemm_bf16_nt<<<dim3(DKV / 128, S_LEN / 128), blk256, 0, stream>>>(
        x_bf, wk_bf, kf, S_LEN, DKV, DMODEL);
    gemm_bf16_nt<<<dim3(DKV / 128, S_LEN / 128), blk256, 0, stream>>>(
        x_bf, wv_bf, vf, S_LEN, DKV, DMODEL);

    // 2) RoPE + transpose + pack; emits k_new/v_new f32 outputs
    float* out_k = out + (size_t)S_LEN * DMODEL;
    float* out_v = out_k + (size_t)HKV * S_LEN * HD;
    rope_pack_kernel<<<dim3(S_LEN, HQ + 2 * HKV), dim3(HD), 0, stream>>>(
        qf, kf, vf, freq, q_bf, k_bf, v_bf, out_k, out_v);

    // 3) causal flash attention (GQA), bf16 output
    flash_attn_kernel<<<dim3(S_LEN / 64, HQ), dim3(128), 0, stream>>>(
        q_bf, k_bf, v_bf, attn_bf);

    // 4) output projection
    gemm_bf16_nt<<<dim3(DMODEL / 128, S_LEN / 128), blk256, 0, stream>>>(
        attn_bf, wo_bf, out, S_LEN, DMODEL, DMODEL);
}
